// CascadeTwoStreamNet_78503412236443
// MI455X (gfx1250) — compile-verified, matrix operand order checked by reading, not run on hardware
//
#include <hip/hip_runtime.h>
#include <hip/hip_bf16.h>
#include <math.h>

// ---------------------------------------------------------------------------
// CDNA5 (gfx1250) implementation of CascadeTwoStreamNet forward pass.
// All dense layers go through one wave32 WMMA GEMM kernel
// (v_wmma_f32_16x16x32_f16, f32 accumulate) with a branch-free, vectorized
// inner loop: A rows are zero-padded to Kp (multiple of 32) and loaded as
// float4s; weights are pre-transposed/converted to f16 [N][Kp] so the B
// operand is two aligned 16-byte loads per lane per k-step.
// ---------------------------------------------------------------------------

typedef __attribute__((ext_vector_type(16))) _Float16 v16h;
typedef __attribute__((ext_vector_type(8)))  _Float16 v8h;
typedef __attribute__((ext_vector_type(8)))  float    v8f;

constexpr int   kB   = 8;
constexpr int   kN   = 16384;
constexpr int   kHID = 64;
constexpr int   kMID = 512;
constexpr int   kANC = 128;
constexpr int   kKST = 32;
constexpr int   kKGN = 20;
constexpr float kR2  = 1.0f;
constexpr float kBNS = 0.9999950000374997f;   // 1/sqrt(1+1e-5)
constexpr float kINF = 3.402823466e38f;

// Epilogue modes
#define EP_BIAS      0
#define EP_BIAS_RELU 1
#define EP_RELU_BN   2   // bn(relu(x+b))    (stem / upm / upd)
#define EP_BN_RELU   3   // relu(bn(x+b))    (edge_conv / fus)

// ---------------------------------------------------------------------------
// Weight prep: Wt[n*Kp + k] = (f16) W[k*N + n], zero-padded k >= K.
// ---------------------------------------------------------------------------
__global__ void k_prep_w(const float* __restrict__ W, int K, int N,
                         _Float16* __restrict__ Wt, int Kp)
{
    const int i = blockIdx.x * 256 + threadIdx.x;
    if (i >= N * Kp) return;
    const int n = i / Kp, k = i % Kp;
    Wt[i] = (k < K) ? (_Float16)W[(size_t)k * N + n] : (_Float16)0.f;
}

// ---------------------------------------------------------------------------
// WMMA GEMM: out[M,Ncols] = epilogue(A[M,Kp] @ Wt[Ncols,Kp]^T)
// One wave per 16x16 tile. Requirements (guaranteed by the launcher):
//   - M, Ncols multiples of 16; Kp multiple of 32
//   - A rows zero-padded to lda >= Kp, lda multiple of 8, 16B-aligned base
//   - Wt is f16, row n holds K-contiguous (padded) weights for output col n
// A-operand lane layout per ISA 7.12.2: lane l holds row (l&15); element e
// of v16h maps to k = e + (e&8) + 8*(l>>4)  -> two contiguous 8-wide blocks.
// ---------------------------------------------------------------------------
__global__ void k_gemm_wmma(const float* __restrict__ A, int lda,
                            const _Float16* __restrict__ Wt, int Kp,
                            const float* __restrict__ bias,
                            const float* __restrict__ gam,
                            const float* __restrict__ bet,
                            float* __restrict__ out, int ldo,
                            int mode)
{
    const int lane = threadIdx.x;       // 0..31
    const int m0   = blockIdx.x * 16;
    const int n0   = blockIdx.y * 16;
    const int hi   = lane >> 4;         // k-half select
    const int lo   = lane & 15;         // row (A) / col (B)

    const float*    arow = A  + (size_t)(m0 + lo) * lda + hi * 8;
    const _Float16* brow = Wt + (size_t)(n0 + lo) * Kp  + hi * 8;

    v8f acc = {};
    for (int k0 = 0; k0 < Kp; k0 += 32) {
        const float4 a0 = *(const float4*)(arow + k0);
        const float4 a1 = *(const float4*)(arow + k0 + 4);
        const float4 a2 = *(const float4*)(arow + k0 + 16);
        const float4 a3 = *(const float4*)(arow + k0 + 20);
        const v8h    b0 = *(const v8h*)(brow + k0);
        const v8h    b1 = *(const v8h*)(brow + k0 + 16);
        v16h av;
        av[0]  = (_Float16)a0.x; av[1]  = (_Float16)a0.y;
        av[2]  = (_Float16)a0.z; av[3]  = (_Float16)a0.w;
        av[4]  = (_Float16)a1.x; av[5]  = (_Float16)a1.y;
        av[6]  = (_Float16)a1.z; av[7]  = (_Float16)a1.w;
        av[8]  = (_Float16)a2.x; av[9]  = (_Float16)a2.y;
        av[10] = (_Float16)a2.z; av[11] = (_Float16)a2.w;
        av[12] = (_Float16)a3.x; av[13] = (_Float16)a3.y;
        av[14] = (_Float16)a3.z; av[15] = (_Float16)a3.w;
        const v16h bv = __builtin_shufflevector(b0, b1,
                            0,1,2,3,4,5,6,7,8,9,10,11,12,13,14,15);
        acc = __builtin_amdgcn_wmma_f32_16x16x32_f16(
                  false, av, false, bv, (short)0, acc, false, false);
    }

    const int n = n0 + lo;
    const float bsv = bias ? bias[n] : 0.f;
    float gs = 0.f, bbv = 0.f;
    if (mode >= EP_RELU_BN) { gs = gam[n] * kBNS; bbv = bet[n]; }
#pragma unroll
    for (int v = 0; v < 8; ++v) {
        const int m = m0 + v + hi * 8;
        float x = acc[v] + bsv;
        if      (mode == EP_BIAS_RELU) x = fmaxf(x, 0.f);
        else if (mode == EP_RELU_BN)   { x = fmaxf(x, 0.f); x = x * gs + bbv; }
        else if (mode == EP_BN_RELU)   { x = x * gs + bbv;  x = fmaxf(x, 0.f); }
        out[(size_t)m * ldo + n] = x;
    }
}

// ---------------------------------------------------------------------------
// Input BN + raw position extraction
// ---------------------------------------------------------------------------
__global__ void k_input_bn(const float* __restrict__ x,
                           const float* __restrict__ g,
                           const float* __restrict__ b,
                           float* __restrict__ norm_x,
                           float* __restrict__ raw_pos)
{
    const int i = blockIdx.x * 256 + threadIdx.x;
    if (i >= kB * kN) return;
#pragma unroll
    for (int c = 0; c < 4; ++c) {
        const float xv = x[(size_t)i * 4 + c];
        norm_x[(size_t)i * 4 + c] = xv * (g[c] * kBNS) + b[c];
        if (c < 3) raw_pos[(size_t)i * 3 + c] = xv;
    }
}

// Deterministic mid-point sampling (coprime-stride pseudo-permutation)
__global__ void k_sample_mid(const float* __restrict__ raw_pos,
                             float* __restrict__ pos_mid)
{
    const int i = blockIdx.x * 256 + threadIdx.x;
    if (i >= kB * kMID) return;
    const int b = i / kMID, m = i % kMID;
    const int idx = (int)(((unsigned)(m * 2741 + b * 4099 + 17)) & (kN - 1));
#pragma unroll
    for (int c = 0; c < 3; ++c)
        pos_mid[(size_t)i * 3 + c] = raw_pos[((size_t)b * kN + idx) * 3 + c];
}

// ---------------------------------------------------------------------------
// Stem kNN: top-32 nearest of 16384 pts per (b,mid). One 256-thread block per
// row; 32 passes of block-wide argmin; per-thread 64-bit used-mask.
// ---------------------------------------------------------------------------
__global__ void k_stem_knn(const float* __restrict__ pos_mid,
                           const float* __restrict__ raw_pos,
                           int*   __restrict__ nbr,
                           float* __restrict__ nbrd2)
{
    const int bm = blockIdx.x;                 // 0..B*MID-1
    const int b  = bm / kMID;
    const float* pts = raw_pos + (size_t)b * kN * 3;
    const float qx = pos_mid[(size_t)bm * 3 + 0];
    const float qy = pos_mid[(size_t)bm * 3 + 1];
    const float qz = pos_mid[(size_t)bm * 3 + 2];
    const int t = threadIdx.x;                 // 0..255
    const int CH = kN / 256;                   // 64 pts per thread
    const int base = t * CH;
    unsigned long long used = 0ull;

    __shared__ float s_val[256];
    __shared__ int   s_idx[256];
    __shared__ int   s_win;

    for (int kk = 0; kk < kKST; ++kk) {
        float bestd = kINF; int besti = base;
        for (int j = 0; j < CH; ++j) {
            if ((used >> j) & 1ull) continue;
            const int p = base + j;
            const float dx = pts[(size_t)p * 3 + 0] - qx;
            const float dy = pts[(size_t)p * 3 + 1] - qy;
            const float dz = pts[(size_t)p * 3 + 2] - qz;
            const float d = dx * dx + dy * dy + dz * dz;
            if (d < bestd) { bestd = d; besti = p; }
        }
        s_val[t] = bestd; s_idx[t] = besti;
        __syncthreads();
        for (int s2 = 128; s2 > 0; s2 >>= 1) {
            if (t < s2 && s_val[t + s2] < s_val[t]) {
                s_val[t] = s_val[t + s2]; s_idx[t] = s_idx[t + s2];
            }
            __syncthreads();
        }
        if (t == 0) {
            s_win = s_idx[0];
            nbr[(size_t)bm * kKST + kk]   = s_idx[0];
            nbrd2[(size_t)bm * kKST + kk] = s_val[0];
        }
        __syncthreads();
        const int win = s_win;
        if (win >= base && win < base + CH) used |= 1ull << (win - base);
        __syncthreads();
    }
}

// Stem edge features: [norm_x(nbr) | pos(nbr)-pos_mid | zeros]  (32-padded)
__global__ void k_stem_edge(const float* __restrict__ norm_x,
                            const float* __restrict__ raw_pos,
                            const float* __restrict__ pos_mid,
                            const int*   __restrict__ nbr,
                            float* __restrict__ edge)
{
    const int e = blockIdx.x * 256 + threadIdx.x;
    if (e >= kB * kMID * kKST) return;
    const int bm = e / kKST;
    const int b  = bm / kMID;
    const int j  = nbr[e];
    const size_t gj = (size_t)b * kN + j;
    float* E = edge + (size_t)e * 32;
#pragma unroll
    for (int c = 0; c < 4; ++c) E[c] = norm_x[gj * 4 + c];
#pragma unroll
    for (int c = 0; c < 3; ++c)
        E[4 + c] = raw_pos[gj * 3 + c] - pos_mid[(size_t)bm * 3 + c];
#pragma unroll
    for (int c = 7; c < 32; ++c) E[c] = 0.f;
}

// Masked max over K_STEM neighbors (invalid -> -inf; all-invalid -> 0)
__global__ void k_stem_max(const float* __restrict__ h,
                           const float* __restrict__ nbrd2,
                           float* __restrict__ x_mid)
{
    const int bm = blockIdx.x;
    const int t  = threadIdx.x;   // 64 feature dims
    float best = 0.f; bool any = false;
    for (int k = 0; k < kKST; ++k) {
        if (nbrd2[(size_t)bm * kKST + k] <= kR2) {
            const float v = h[((size_t)bm * kKST + k) * kHID + t];
            best = any ? fmaxf(best, v) : v; any = true;
        }
    }
    x_mid[(size_t)bm * kHID + t] = any ? best : 0.f;
}

// Feature-space distance matrix d2f[b][i][j]
__global__ void k_d2f(const float* __restrict__ x_mid,
                      float* __restrict__ d2f)
{
    const int bi = blockIdx.x;             // 0..B*MID-1
    const int b  = bi / kMID;
    __shared__ float si[kHID];
    const int t = threadIdx.x;             // 256
    if (t < kHID) si[t] = x_mid[(size_t)bi * kHID + t];
    __syncthreads();
    for (int j = t; j < kMID; j += 256) {
        const float* rj = x_mid + ((size_t)b * kMID + j) * kHID;
        float s = 0.f;
        for (int c = 0; c < kHID; ++c) { const float d = si[c] - rj[c]; s += d * d; }
        d2f[(size_t)bi * kMID + j] = s;
    }
}

// top-20 smallest per row (one thread per row)
__global__ void k_topk_gnn(const float* __restrict__ d2f,
                           int* __restrict__ nn_idx)
{
    const int r = blockIdx.x * 128 + threadIdx.x;
    if (r >= kB * kMID) return;
    const float* row = d2f + (size_t)r * kMID;
    int picked[kKGN];
    for (int k = 0; k < kKGN; ++k) {
        float best = kINF; int bi = 0;
        for (int j = 0; j < kMID; ++j) {
            bool skip = false;
            for (int q = 0; q < k; ++q) if (picked[q] == j) { skip = true; break; }
            if (skip) continue;
            const float d = row[j];
            if (d < best) { best = d; bi = j; }
        }
        picked[k] = bi;
        nn_idx[(size_t)r * kKGN + k] = bi;
    }
}

// EdgeConv input features m = [xi | xj - xi]   (128 dims)
__global__ void k_gnn_edge(const float* __restrict__ x_mid,
                           const int*   __restrict__ nn_idx,
                           float* __restrict__ m_feat)
{
    const size_t tid = (size_t)blockIdx.x * 256 + threadIdx.x;
    const size_t total = (size_t)kB * kMID * kKGN * (2 * kHID);
    if (tid >= total) return;
    const int c = (int)(tid & 127);
    const size_t r = tid >> 7;              // (b*MID+i)*KGN + kk
    const size_t bi = r / kKGN;             // b*MID + i
    const int b = (int)(bi / kMID);
    const int j = nn_idx[r];
    const size_t xi = bi * kHID;
    const size_t xj = ((size_t)b * kMID + j) * kHID;
    float v;
    if (c < kHID) v = x_mid[xi + c];
    else          v = x_mid[xj + (c - kHID)] - x_mid[xi + (c - kHID)];
    m_feat[tid] = v;
}

// max over K_GNN neighbors, write into a column slice of `out`
__global__ void k_maxk(const float* __restrict__ h,
                       float* __restrict__ out, int ldo, int coloff)
{
    const int bm = blockIdx.x;
    const int t  = threadIdx.x;     // 64
    float best = -kINF;
    for (int k = 0; k < kKGN; ++k)
        best = fmaxf(best, h[((size_t)bm * kKGN + k) * kHID + t]);
    out[(size_t)bm * ldo + coloff + t] = best;
}

__global__ void k_residual(float* __restrict__ x, const float* __restrict__ r, int n)
{
    const int i = blockIdx.x * 256 + threadIdx.x;
    if (i < n) x[i] += r[i];
}

// Column std (ddof=1) over all B*MID rows: cols 0-2 = pos_mid, 3-66 = x_mid
__global__ void k_colstd(const float* __restrict__ pos_mid,
                         const float* __restrict__ x_mid,
                         float* __restrict__ stds)
{
    const int c = blockIdx.x;     // 0..66
    const int t = threadIdx.x;    // 256
    const int rows = kB * kMID;
    const float* src; int ld, col;
    if (c < 3) { src = pos_mid; ld = 3; col = c; }
    else       { src = x_mid;  ld = kHID; col = c - 3; }
    float s = 0.f, ss = 0.f;
    for (int r = t; r < rows; r += 256) {
        const float v = src[(size_t)r * ld + col];
        s += v; ss += v * v;
    }
    __shared__ float rs[256], rss[256];
    rs[t] = s; rss[t] = ss;
    __syncthreads();
    for (int s2 = 128; s2 > 0; s2 >>= 1) {
        if (t < s2) { rs[t] += rs[t + s2]; rss[t] += rss[t + s2]; }
        __syncthreads();
    }
    if (t == 0) {
        const float mean = rs[0] / rows;
        const float var  = (rss[0] - rs[0] * mean) / (rows - 1);
        stds[c] = sqrtf(fmaxf(var, 0.f));
    }
}

__global__ void k_metric(const float* __restrict__ pos_mid,
                         const float* __restrict__ x_mid,
                         const float* __restrict__ stds,
                         float* __restrict__ metric)
{
    const int i = blockIdx.x * 256 + threadIdx.x;
    if (i >= kB * kMID * 67) return;
    const int c = i % 67;
    const int r = i / 67;
    float v;
    if (c < 3) v = pos_mid[(size_t)r * 3 + c] * 1.5f / (stds[c] + 1e-6f);
    else       v = x_mid[(size_t)r * kHID + (c - 3)] / (stds[c] + 1e-6f);
    metric[i] = v;
}

// Farthest-point sampling, one 256-thread block per batch
__global__ void k_fps(const float* __restrict__ metric,
                      int* __restrict__ idx_anchor)
{
    const int b = blockIdx.x;
    const float* M_ = metric + (size_t)b * kMID * 67;
    __shared__ float d2min[kMID];
    __shared__ float s_val[256];
    __shared__ int   s_idx[256];
    __shared__ int   s_last;
    const int t = threadIdx.x;
    for (int p = t; p < kMID; p += 256) d2min[p] = kINF;
    if (t == 0) s_last = 0;
    __syncthreads();
    for (int it = 0; it < kANC; ++it) {
        const int last = s_last;
        if (t == 0) idx_anchor[(size_t)b * kANC + it] = last;
        const float* lp = M_ + (size_t)last * 67;
        float bv = -1.f; int bi = 0;
        for (int p = t; p < kMID; p += 256) {
            const float* pp = M_ + (size_t)p * 67;
            float s = 0.f;
            for (int c = 0; c < 67; ++c) { const float d = pp[c] - lp[c]; s += d * d; }
            const float m = fminf(d2min[p], s);
            d2min[p] = m;
            if (m > bv) { bv = m; bi = p; }
        }
        s_val[t] = bv; s_idx[t] = bi;
        __syncthreads();
        for (int s2 = 128; s2 > 0; s2 >>= 1) {
            if (t < s2 && s_val[t + s2] > s_val[t]) {
                s_val[t] = s_val[t + s2]; s_idx[t] = s_idx[t + s2];
            }
            __syncthreads();
        }
        if (t == 0) s_last = s_idx[0];
        __syncthreads();
    }
}

// Gather anchors -> gi_in = [x_anchor | bn(pos_anchor) | zeros] (96-padded)
__global__ void k_anchor_gather(const int* __restrict__ idx_anchor,
                                const float* __restrict__ pos_mid,
                                const float* __restrict__ x_mid,
                                const float* __restrict__ pg,
                                const float* __restrict__ pb,
                                float* __restrict__ gi_in,
                                float* __restrict__ pos_anchor)
{
    const int i = blockIdx.x * 256 + threadIdx.x;
    if (i >= kB * kANC) return;
    const int b = i / kANC;
    const int a = idx_anchor[i];
    const size_t src = (size_t)b * kMID + a;
#pragma unroll 8
    for (int c = 0; c < kHID; ++c)
        gi_in[(size_t)i * 96 + c] = x_mid[src * kHID + c];
#pragma unroll
    for (int c = 0; c < 3; ++c) {
        const float p = pos_mid[src * 3 + c];
        pos_anchor[(size_t)i * 3 + c] = p;
        gi_in[(size_t)i * 96 + kHID + c] = p * (pg[c] * kBNS) + pb[c];
    }
#pragma unroll
    for (int c = 67; c < 96; ++c) gi_in[(size_t)i * 96 + c] = 0.f;
}

// 4-head attention over 128 anchors; online softmax; one block per (b,h)
__global__ void k_attn(const float* __restrict__ qkv,
                       float* __restrict__ o)
{
    const int bh = blockIdx.x;
    const int b = bh / 4, h = bh % 4;
    const int i = threadIdx.x;         // 0..127 query index
    __shared__ float sk[kANC][16];
    __shared__ float sv[kANC][16];
    const size_t row = ((size_t)b * kANC + i) * 192;
#pragma unroll
    for (int d = 0; d < 16; ++d) {
        sk[i][d] = qkv[row + 64  + h * 16 + d];
        sv[i][d] = qkv[row + 128 + h * 16 + d];
    }
    __syncthreads();
    float q[16];
#pragma unroll
    for (int d = 0; d < 16; ++d) q[d] = qkv[row + h * 16 + d];

    float m = -kINF, l = 0.f, acc[16];
#pragma unroll
    for (int d = 0; d < 16; ++d) acc[d] = 0.f;
    for (int j = 0; j < kANC; ++j) {
        float s = 0.f;
#pragma unroll
        for (int d = 0; d < 16; ++d) s += q[d] * sk[j][d];
        s *= 0.25f;                          // 1/sqrt(16)
        const float nm = fmaxf(m, s);
        const float corr = __expf(m - nm);
        const float p = __expf(s - nm);
        l = l * corr + p;
#pragma unroll
        for (int d = 0; d < 16; ++d) acc[d] = acc[d] * corr + p * sv[j][d];
        m = nm;
    }
    const float inv = 1.f / l;
#pragma unroll
    for (int d = 0; d < 16; ++d)
        o[((size_t)b * kANC + i) * kHID + h * 16 + d] = acc[d] * inv;
}

// out = LayerNorm(a + b2) * g + bb   (rows of 64; one 64-thread block per row)
__global__ void k_add_ln(const float* __restrict__ a,
                         const float* __restrict__ b2,
                         const float* __restrict__ g,
                         const float* __restrict__ bb,
                         float* __restrict__ out)
{
    const int r = blockIdx.x;
    const int t = threadIdx.x;   // 64
    __shared__ float red[64];
    const float v = a[(size_t)r * kHID + t] + b2[(size_t)r * kHID + t];
    red[t] = v;
    __syncthreads();
    for (int s2 = 32; s2 > 0; s2 >>= 1) {
        if (t < s2) red[t] += red[t + s2];
        __syncthreads();
    }
    const float mean = red[0] * (1.f / kHID);
    __syncthreads();
    const float dv = v - mean;
    red[t] = dv * dv;
    __syncthreads();
    for (int s2 = 32; s2 > 0; s2 >>= 1) {
        if (t < s2) red[t] += red[t + s2];
        __syncthreads();
    }
    const float var = red[0] * (1.f / kHID);
    out[(size_t)r * kHID + t] = dv * rsqrtf(var + 1e-5f) * g[t] + bb[t];
}

// kNN-3 interpolation anchors -> mid; writes fuse_in = [x_mid | ctx_mid]
__global__ void k_interp_a2m(const float* __restrict__ pos_mid,
                             const float* __restrict__ pos_anchor,
                             const float* __restrict__ xg,
                             const float* __restrict__ x_mid,
                             float* __restrict__ fuse_in)
{
    const int bm = blockIdx.x;          // b*MID + i
    const int b  = bm / kMID;
    const int t  = threadIdx.x;         // 64
    __shared__ int   sj[3];
    __shared__ float sw[3];
    __shared__ float swsum;
    if (t == 0) {
        const float qx = pos_mid[(size_t)bm * 3 + 0];
        const float qy = pos_mid[(size_t)bm * 3 + 1];
        const float qz = pos_mid[(size_t)bm * 3 + 2];
        float bd[3] = { kINF, kINF, kINF }; int bi[3] = { 0, 0, 0 };
        for (int j = 0; j < kANC; ++j) {
            const float* p = pos_anchor + ((size_t)b * kANC + j) * 3;
            const float dx = p[0] - qx, dy = p[1] - qy, dz = p[2] - qz;
            const float d = dx * dx + dy * dy + dz * dz;
            if (d < bd[0]) { bd[2]=bd[1]; bi[2]=bi[1]; bd[1]=bd[0]; bi[1]=bi[0]; bd[0]=d; bi[0]=j; }
            else if (d < bd[1]) { bd[2]=bd[1]; bi[2]=bi[1]; bd[1]=d; bi[1]=j; }
            else if (d < bd[2]) { bd[2]=d; bi[2]=j; }
        }
        float ws = 0.f;
        for (int q = 0; q < 3; ++q) {
            const float w = 1.f / fmaxf(bd[q], 1e-16f);
            sw[q] = w; sj[q] = bi[q]; ws += w;
        }
        swsum = ws;
    }
    __syncthreads();
    float acc = 0.f;
#pragma unroll
    for (int q = 0; q < 3; ++q)
        acc += sw[q] * xg[((size_t)b * kANC + sj[q]) * kHID + t];
    fuse_in[(size_t)bm * 128 + t]        = x_mid[(size_t)bm * kHID + t];
    fuse_in[(size_t)bm * 128 + kHID + t] = acc / swsum;
}

// kNN-3 interpolation mid -> dense; writes di = [norm_x | ctx | zeros] (96)
__global__ void k_interp_m2d(const float* __restrict__ raw_pos,
                             const float* __restrict__ pos_mid,
                             const float* __restrict__ xf,
                             const float* __restrict__ norm_x,
                             float* __restrict__ di)
{
    const int bn = blockIdx.x;          // b*N + n
    const int b  = bn / kN;
    const int t  = threadIdx.x;         // 64
    __shared__ int   sj[3];
    __shared__ float sw[3];
    __shared__ float swsum;
    if (t == 0) {
        const float qx = raw_pos[(size_t)bn * 3 + 0];
        const float qy = raw_pos[(size_t)bn * 3 + 1];
        const float qz = raw_pos[(size_t)bn * 3 + 2];
        float bd[3] = { kINF, kINF, kINF }; int bi[3] = { 0, 0, 0 };
        for (int j = 0; j < kMID; ++j) {
            const float* p = pos_mid + ((size_t)b * kMID + j) * 3;
            const float dx = p[0] - qx, dy = p[1] - qy, dz = p[2] - qz;
            const float d = dx * dx + dy * dy + dz * dz;
            if (d < bd[0]) { bd[2]=bd[1]; bi[2]=bi[1]; bd[1]=bd[0]; bi[1]=bi[0]; bd[0]=d; bi[0]=j; }
            else if (d < bd[1]) { bd[2]=bd[1]; bi[2]=bi[1]; bd[1]=d; bi[1]=j; }
            else if (d < bd[2]) { bd[2]=d; bi[2]=j; }
        }
        float ws = 0.f;
        for (int q = 0; q < 3; ++q) {
            const float w = 1.f / fmaxf(bd[q], 1e-16f);
            sw[q] = w; sj[q] = bi[q]; ws += w;
        }
        swsum = ws;
    }
    __syncthreads();
    float acc = 0.f;
#pragma unroll
    for (int q = 0; q < 3; ++q)
        acc += sw[q] * xf[((size_t)b * kMID + sj[q]) * kHID + t];
    di[(size_t)bn * 96 + 4 + t] = acc / swsum;
    if (t < 4)  di[(size_t)bn * 96 + t] = norm_x[(size_t)bn * 4 + t];
    if (t < 28) di[(size_t)bn * 96 + 68 + t] = 0.f;
}

// Final 32 -> 1 projection
__global__ void k_final(const float* __restrict__ z2,
                        const float* __restrict__ w3,
                        const float* __restrict__ b3,
                        float* __restrict__ out)
{
    const int i = blockIdx.x * 256 + threadIdx.x;
    if (i >= kB * kN) return;
    float s = b3[0];
#pragma unroll
    for (int c = 0; c < 32; ++c) s += z2[(size_t)i * 32 + c] * w3[c];
    out[i] = s;
}

// ---------------------------------------------------------------------------
// Host side
// ---------------------------------------------------------------------------
static void gemm(hipStream_t s, const float* A, int lda,
                 const _Float16* Wt, int Kp,
                 const float* bias, const float* g, const float* bb,
                 float* out, int ldo, int M, int Nc, int mode)
{
    dim3 gr((unsigned)(M / 16), (unsigned)(Nc / 16));
    k_gemm_wmma<<<gr, dim3(32), 0, s>>>(A, lda, Wt, Kp, bias, g, bb,
                                        out, ldo, mode);
}

extern "C" void kernel_launch(void* const* d_in, const int* in_sizes, int n_in,
                              void* d_out, int out_size, void* d_ws, size_t ws_size,
                              hipStream_t stream)
{
    (void)in_sizes; (void)n_in; (void)out_size; (void)ws_size;

    const float* x_in = (const float*)d_in[0];
    // d_in[1] = batch indices (unused)
    int pi = 2;
    // JAX pytree leaves: dict keys in sorted order at every level.
    const float* fus_b   = (const float*)d_in[pi++];
    const float* fus_bn  = (const float*)d_in[pi++];
    const float* fus_g   = (const float*)d_in[pi++];
    const float* fus_w   = (const float*)d_in[pi++];
    const float* inbn_b  = (const float*)d_in[pi++];
    const float* inbn_g  = (const float*)d_in[pi++];
    const float* posbn_b = (const float*)d_in[pi++];
    const float* posbn_g = (const float*)d_in[pi++];
    const float* pre_b   = (const float*)d_in[pi++];
    const float* pre_w   = (const float*)d_in[pi++];
    const float* sp_b1   = (const float*)d_in[pi++];
    const float* sp_b1n  = (const float*)d_in[pi++];
    const float* sp_b2   = (const float*)d_in[pi++];
    const float* sp_b2n  = (const float*)d_in[pi++];
    const float* sp_g1   = (const float*)d_in[pi++];
    const float* sp_g2   = (const float*)d_in[pi++];
    const float* sp_w1   = (const float*)d_in[pi++];
    const float* sp_w2   = (const float*)d_in[pi++];
    const float* st_b1   = (const float*)d_in[pi++];
    const float* st_b1n  = (const float*)d_in[pi++];
    const float* st_b2   = (const float*)d_in[pi++];
    const float* st_b2n  = (const float*)d_in[pi++];
    const float* st_g1   = (const float*)d_in[pi++];
    const float* st_g2   = (const float*)d_in[pi++];
    const float* st_w1   = (const float*)d_in[pi++];
    const float* st_w2   = (const float*)d_in[pi++];
    const float* tp_b1   = (const float*)d_in[pi++];
    const float* tp_b1n  = (const float*)d_in[pi++];
    const float* tp_b2   = (const float*)d_in[pi++];
    const float* tp_b2n  = (const float*)d_in[pi++];
    const float* tp_g1   = (const float*)d_in[pi++];
    const float* tp_g2   = (const float*)d_in[pi++];
    const float* tp_w1   = (const float*)d_in[pi++];
    const float* tp_w2   = (const float*)d_in[pi++];
    const float* tr_b1   = (const float*)d_in[pi++];
    const float* tr_b2   = (const float*)d_in[pi++];
    const float* tr_bo   = (const float*)d_in[pi++];
    const float* tr_bqkv = (const float*)d_in[pi++];
    const float* tr_ln1b = (const float*)d_in[pi++];
    const float* tr_ln1g = (const float*)d_in[pi++];
    const float* tr_ln2b = (const float*)d_in[pi++];
    const float* tr_ln2g = (const float*)d_in[pi++];
    const float* tr_w1   = (const float*)d_in[pi++];
    const float* tr_w2   = (const float*)d_in[pi++];
    const float* tr_wo   = (const float*)d_in[pi++];
    const float* tr_wqkv = (const float*)d_in[pi++];
    const float* ud_b1   = (const float*)d_in[pi++];
    const float* ud_b1n  = (const float*)d_in[pi++];
    const float* ud_b2   = (const float*)d_in[pi++];
    const float* ud_b2n  = (const float*)d_in[pi++];
    const float* ud_b3   = (const float*)d_in[pi++];
    const float* ud_g1   = (const float*)d_in[pi++];
    const float* ud_g2   = (const float*)d_in[pi++];
    const float* ud_w1   = (const float*)d_in[pi++];
    const float* ud_w2   = (const float*)d_in[pi++];
    const float* ud_w3   = (const float*)d_in[pi++];
    const float* um_b    = (const float*)d_in[pi++];
    const float* um_bn   = (const float*)d_in[pi++];
    const float* um_g    = (const float*)d_in[pi++];
    const float* um_w    = (const float*)d_in[pi++];

    // ---- workspace carve-up (floats) ----
    float* wsf = (float*)d_ws;
    size_t off = 0;
    auto alloc = [&](size_t n) { float* r = wsf + off; off += n; return r; };

    float* raw_pos  = alloc((size_t)kB * kN * 3);
    float* norm_x   = alloc((size_t)kB * kN * 4);
    float* pos_mid  = alloc((size_t)kB * kMID * 3);
    int*   nbr      = (int*)alloc((size_t)kB * kMID * kKST);
    float* nbrd2    = alloc((size_t)kB * kMID * kKST);
    float* edge     = alloc((size_t)kB * kMID * kKST * 32);          // Kp=32
    // region A (reused): stem h1 (8.4M) -> m_feat (10.5M) -> di 96-pad (12.6M)
    float* regA     = alloc((size_t)kB * kN * 96);                   // 12582912
    // region B (reused): stem h2 (8.4M) -> e1 (5.2M) -> z1 (8.4M)
    float* regB     = alloc((size_t)kB * kN * kHID);                 // 8388608
    // region C (reused): e2 (5.2M) -> z2 (4.2M)
    float* regC     = alloc((size_t)kB * kMID * kKGN * kHID);        // 5242880
    float* x_mid    = alloc((size_t)kB * kMID * kHID);
    float* d2f      = alloc((size_t)kB * kMID * kMID);
    int*   nn_idx   = (int*)alloc((size_t)kB * kMID * kKGN);
    float* comb     = alloc((size_t)kB * kMID * 2 * kHID);
    float* refined  = alloc((size_t)kB * kMID * kHID);
    float* stds     = alloc(128);
    float* metric   = alloc((size_t)kB * kMID * 67 + 4);
    int*   idx_anc  = (int*)alloc((size_t)kB * kANC);
    float* pos_anc  = alloc((size_t)kB * kANC * 3 + 4);
    float* gi_in    = alloc((size_t)kB * kANC * 96);                 // Kp=96
    float* gi       = alloc((size_t)kB * kANC * kHID);
    float* qkv      = alloc((size_t)kB * kANC * 3 * kHID);
    float* o_attn   = alloc((size_t)kB * kANC * kHID);
    float* oproj    = alloc((size_t)kB * kANC * kHID);
    float* hh       = alloc((size_t)kB * kANC * kHID);
    float* ff1      = alloc((size_t)kB * kANC * 2 * kHID);
    float* ff2      = alloc((size_t)kB * kANC * kHID);
    float* xglob    = alloc((size_t)kB * kANC * kHID);
    float* fuse_in  = alloc((size_t)kB * kMID * 2 * kHID);
    float* x_fused  = alloc((size_t)kB * kMID * kHID);

    // ---- f16 transposed weight copies (tiny) ----
    _Float16* whbase = (_Float16*)(wsf + off);
    size_t hoff = 0;
    auto alloch = [&](size_t n) { _Float16* r = whbase + hoff; hoff += (n + 7) & ~(size_t)7; return r; };
    _Float16* wt_st1  = alloch(64 * 32);
    _Float16* wt_st2  = alloch(64 * 64);
    _Float16* wt_sp1  = alloch(64 * 128);
    _Float16* wt_sp2  = alloch(64 * 64);
    _Float16* wt_tp1  = alloch(64 * 128);
    _Float16* wt_tp2  = alloch(64 * 64);
    _Float16* wt_fus  = alloch(64 * 128);
    _Float16* wt_pre  = alloch(64 * 96);
    _Float16* wt_qkv  = alloch(192 * 64);
    _Float16* wt_wo   = alloch(64 * 64);
    _Float16* wt_tw1  = alloch(128 * 64);
    _Float16* wt_tw2  = alloch(64 * 128);
    _Float16* wt_um   = alloch(64 * 128);
    _Float16* wt_ud1  = alloch(64 * 96);
    _Float16* wt_ud2  = alloch(32 * 64);

    auto prep = [&](const float* W, int K, int N, _Float16* Wt, int Kp) {
        const int tot = N * Kp;
        k_prep_w<<<(tot + 255) / 256, 256, 0, stream>>>(W, K, N, Wt, Kp);
    };
    prep(st_w1,    7, 64,  wt_st1, 32);
    prep(st_w2,   64, 64,  wt_st2, 64);
    prep(sp_w1,  128, 64,  wt_sp1, 128);
    prep(sp_w2,   64, 64,  wt_sp2, 64);
    prep(tp_w1,  128, 64,  wt_tp1, 128);
    prep(tp_w2,   64, 64,  wt_tp2, 64);
    prep(fus_w,  128, 64,  wt_fus, 128);
    prep(pre_w,   67, 64,  wt_pre, 96);
    prep(tr_wqkv, 64, 192, wt_qkv, 64);
    prep(tr_wo,   64, 64,  wt_wo,  64);
    prep(tr_w1,   64, 128, wt_tw1, 64);
    prep(tr_w2,  128, 64,  wt_tw2, 128);
    prep(um_w,   128, 64,  wt_um,  128);
    prep(ud_w1,   68, 64,  wt_ud1, 96);
    prep(ud_w2,   64, 32,  wt_ud2, 64);

    const int BN  = kB * kN;       // 131072
    const int BM  = kB * kMID;     // 4096
    const int BMK = BM * kKST;     // 131072 stem edges
    const int BMG = BM * kKGN;     // 81920 gnn edges
    const int BA  = kB * kANC;     // 1024

    // ---- 1. input BN, raw positions, mid sampling ----
    k_input_bn<<<(BN + 255) / 256, 256, 0, stream>>>(x_in, inbn_g, inbn_b, norm_x, raw_pos);
    k_sample_mid<<<(BM + 255) / 256, 256, 0, stream>>>(raw_pos, pos_mid);

    // ---- 2. stem kNN + edge MLP + masked max ----
    k_stem_knn<<<BM, 256, 0, stream>>>(pos_mid, raw_pos, nbr, nbrd2);
    k_stem_edge<<<(BMK + 255) / 256, 256, 0, stream>>>(norm_x, raw_pos, pos_mid, nbr, edge);
    gemm(stream, edge, 32,  wt_st1, 32, st_b1, st_g1, st_b1n, regA, kHID, BMK, kHID, EP_RELU_BN);
    gemm(stream, regA, 64,  wt_st2, 64, st_b2, st_g2, st_b2n, regB, kHID, BMK, kHID, EP_RELU_BN);
    k_stem_max<<<BM, kHID, 0, stream>>>(regB, nbrd2, x_mid);

    // ---- 3. GNN: feature kNN + two EdgeConv streams + fusion ----
    k_d2f<<<BM, 256, 0, stream>>>(x_mid, d2f);
    k_topk_gnn<<<(BM + 127) / 128, 128, 0, stream>>>(d2f, nn_idx);
    k_gnn_edge<<<(int)(((size_t)BMG * 128 + 255) / 256), 256, 0, stream>>>(x_mid, nn_idx, regA);
    // spatial stream -> comb[:, 0:64]
    gemm(stream, regA, 128, wt_sp1, 128, sp_b1, sp_g1, sp_b1n, regB, kHID, BMG, kHID, EP_BN_RELU);
    gemm(stream, regB, 64,  wt_sp2, 64,  sp_b2, sp_g2, sp_b2n, regC, kHID, BMG, kHID, EP_BN_RELU);
    k_maxk<<<BM, kHID, 0, stream>>>(regC, comb, 2 * kHID, 0);
    // "temporal" stream -> comb[:, 64:128]
    gemm(stream, regA, 128, wt_tp1, 128, tp_b1, tp_g1, tp_b1n, regB, kHID, BMG, kHID, EP_BN_RELU);
    gemm(stream, regB, 64,  wt_tp2, 64,  tp_b2, tp_g2, tp_b2n, regC, kHID, BMG, kHID, EP_BN_RELU);
    k_maxk<<<BM, kHID, 0, stream>>>(regC, comb, 2 * kHID, kHID);
    // fusion + residual
    gemm(stream, comb, 128, wt_fus, 128, fus_b, fus_g, fus_bn, refined, kHID, BM, kHID, EP_BN_RELU);
    k_residual<<<(BM * kHID + 255) / 256, 256, 0, stream>>>(x_mid, refined, BM * kHID);

    // ---- 4. FPS anchors ----
    k_colstd<<<67, 256, 0, stream>>>(pos_mid, x_mid, stds);
    k_metric<<<(BM * 67 + 255) / 256, 256, 0, stream>>>(pos_mid, x_mid, stds, metric);
    k_fps<<<kB, 256, 0, stream>>>(metric, idx_anc);
    k_anchor_gather<<<(BA + 255) / 256, 256, 0, stream>>>(idx_anc, pos_mid, x_mid,
                                                          posbn_g, posbn_b, gi_in, pos_anc);

    // ---- 5. transformer over anchors ----
    gemm(stream, gi_in, 96, wt_pre, 96, pre_b,   nullptr, nullptr, gi,  kHID,     BA, kHID,     EP_BIAS);
    gemm(stream, gi,    64, wt_qkv, 64, tr_bqkv, nullptr, nullptr, qkv, 3 * kHID, BA, 3 * kHID, EP_BIAS);
    k_attn<<<kB * 4, kANC, 0, stream>>>(qkv, o_attn);
    gemm(stream, o_attn, 64, wt_wo, 64, tr_bo, nullptr, nullptr, oproj, kHID, BA, kHID, EP_BIAS);
    k_add_ln<<<BA, kHID, 0, stream>>>(gi, oproj, tr_ln1g, tr_ln1b, hh);
    gemm(stream, hh,  64,  wt_tw1, 64,  tr_b1, nullptr, nullptr, ff1, 2 * kHID, BA, 2 * kHID, EP_BIAS_RELU);
    gemm(stream, ff1, 128, wt_tw2, 128, tr_b2, nullptr, nullptr, ff2, kHID,     BA, kHID,     EP_BIAS);
    k_add_ln<<<BA, kHID, 0, stream>>>(hh, ff2, tr_ln2g, tr_ln2b, xglob);

    // ---- 6. upsample anchors -> mid ----
    k_interp_a2m<<<BM, kHID, 0, stream>>>(pos_mid, pos_anc, xglob, x_mid, fuse_in);
    gemm(stream, fuse_in, 128, wt_um, 128, um_b, um_g, um_bn, x_fused, kHID, BM, kHID, EP_RELU_BN);

    // ---- 7. upsample mid -> dense + head ----
    k_interp_m2d<<<BN, kHID, 0, stream>>>(raw_pos, pos_mid, x_fused, norm_x, regA);  // regA = di [BN,96]
    gemm(stream, regA, 96, wt_ud1, 96, ud_b1, ud_g1, ud_b1n, regB, kHID, BN, kHID, EP_RELU_BN);
    gemm(stream, regB, 64, wt_ud2, 64, ud_b2, ud_g2, ud_b2n, regC, 32,   BN, 32,   EP_RELU_BN);
    k_final<<<(BN + 255) / 256, 256, 0, stream>>>(regC, ud_w3, ud_b3, (float*)d_out);
}
